// TensplitGAT_26061861552525
// MI455X (gfx1250) — compile-verified, matrix operand order checked by reading, not run on hardware
//
#include <hip/hip_runtime.h>
#include <hip/hip_fp16.h>
#include <math.h>

typedef __attribute__((ext_vector_type(16))) _Float16 v16h;
typedef __attribute__((ext_vector_type(8)))  float    v8f;
typedef __attribute__((ext_vector_type(4)))  float    v4f;

#define NNODES 100000
#define NEDGES 1600000
#define INDIM  256
#define HIDDIM 64
#define NCLS   32

// ---------------- wave32 reductions ----------------
__device__ __forceinline__ float wave_sum(float v) {
#pragma unroll
  for (int off = 16; off > 0; off >>= 1) v += __shfl_xor(v, off, 32);
  return v;
}
__device__ __forceinline__ float wave_max(float v) {
#pragma unroll
  for (int off = 16; off > 0; off >>= 1) v = fmaxf(v, __shfl_xor(v, off, 32));
  return v;
}

// sign-split float atomic max (monotonic int/uint mapping); dest init = -inf
__device__ __forceinline__ void atomicMaxF32(float* addr, float value) {
  if (value >= 0.0f) atomicMax((int*)addr, __float_as_int(value));
  else               atomicMin((unsigned int*)addr, __float_as_uint(value));
}

__device__ __forceinline__ float elu_f(float x) {
  return x > 0.0f ? x : (expf(x) - 1.0f);
}

// ---------------- attention projections: wave per node ----------------
__global__ void proj_kernel(const float* __restrict__ h, const float* __restrict__ attw,
                            float* __restrict__ pa, float* __restrict__ pb) {
  int node = blockIdx.x * (blockDim.x >> 5) + (threadIdx.x >> 5);
  int lane = threadIdx.x & 31;
  if (node >= NNODES) return;
  const float* row = h + (size_t)node * HIDDIM;
  float x0 = row[lane], x1 = row[lane + 32];
  float a = x0 * attw[lane]          + x1 * attw[lane + 32];
  float b = x0 * attw[HIDDIM + lane] + x1 * attw[HIDDIM + lane + 32];
  a = wave_sum(a); b = wave_sum(b);
  if (lane == 0) { pa[node] = a; pb[node] = b; }
}

__global__ void init_rows(float* __restrict__ rowmax, float* __restrict__ rowsum) {
  int i = blockIdx.x * blockDim.x + threadIdx.x;
  if (i < NNODES) { rowmax[i] = -__builtin_inff(); rowsum[i] = 0.0f; }
}

__global__ void zero_kernel(float* __restrict__ p, int n) {
  int i = blockIdx.x * blockDim.x + threadIdx.x;
  if (i < n) p[i] = 0.0f;
}

// ---------------- edge softmax ----------------
__global__ void edge_score(const int* __restrict__ src, const int* __restrict__ dst,
                           const float* __restrict__ pa, const float* __restrict__ pb,
                           float* __restrict__ att, float* __restrict__ rowmax) {
  int e = blockIdx.x * blockDim.x + threadIdx.x;
  if (e >= NEDGES) return;
  float x = pa[src[e]] + pb[dst[e]];
  float sc = x > 0.0f ? x : 0.01f * x;   // leaky_relu, slope 0.01
  att[e] = sc;
  atomicMaxF32(&rowmax[src[e]], sc);
}

__global__ void edge_exp(const int* __restrict__ src, float* __restrict__ att,
                         const float* __restrict__ rowmax, float* __restrict__ rowsum) {
  int e = blockIdx.x * blockDim.x + threadIdx.x;
  if (e >= NEDGES) return;
  int s = src[e];
  float ex = expf(att[e] - rowmax[s]);
  att[e] = ex;
  atomicAdd(&rowsum[s], ex);
}

__global__ void edge_norm(const int* __restrict__ src, float* __restrict__ att,
                          const float* __restrict__ rowsum) {
  int e = blockIdx.x * blockDim.x + threadIdx.x;
  if (e >= NEDGES) return;
  att[e] = att[e] / rowsum[src[e]];
}

// ---------------- weight pre-pack: f32 W[K x NC] -> f16 WMMA B fragments ----------
// Fragment f = tn*(K/32) + kc. Per lane l: col n = tn*16 + (l&15), hi = l>>4,
// element e -> k = kc*32 + ((e>>3)<<4) + hi*8 + (e&7). Stored contiguously:
// Bp[(f*32 + l)*16 .. +15] so the GEMM loads one aligned 32B v16h per fragment.
template<int K, int NC>
__global__ void pack_w(const float* __restrict__ W, _Float16* __restrict__ Bp) {
  constexpr int KC = K / 32, TN = NC / 16, FR = KC * TN;
  int t = blockIdx.x * blockDim.x + threadIdx.x;
  if (t >= FR * 32) return;
  int frag = t >> 5, lane = t & 31;
  int tn = frag / KC, kc = frag % KC;
  int hi = lane >> 4, n = tn * 16 + (lane & 15);
  int kb = kc * 32 + hi * 8;
  v16h b;
#pragma unroll
  for (int e = 0; e < 8; ++e) {
    b[e]     = (_Float16)W[(size_t)(kb + e) * NC + n];
    b[8 + e] = (_Float16)W[(size_t)(kb + 16 + e) * NC + n];
  }
  *(v16h*)(Bp + (size_t)t * 16) = b;
}

// ---------------- dense GEMM via V_WMMA_F32_16X16X32_F16 ----------------
// One wave per 16-row strip: loads each A fragment once, multiplies against all
// NC/16 pre-packed B fragments (NC/16 accumulators). Optionally applies ELU to
// A elements during the f32->f16 conversion (fusing the previous layer's
// activation into this kernel's streaming read).
template<int K, int NC, bool ELU_A>
__global__ void gemm_wmma(const float* __restrict__ A, const _Float16* __restrict__ Bp,
                          float* __restrict__ Y) {
  constexpr int KC = K / 32, TN = NC / 16;
  int tm = blockIdx.x * (blockDim.x >> 5) + (threadIdx.x >> 5);
  if (tm >= NNODES / 16) return;          // whole-wave uniform: EXEC stays all-ones
  int lane = threadIdx.x & 31;
  int ml = lane & 15, hi = lane >> 4;
  const float* arow = A + (size_t)(tm * 16 + ml) * K;
  const v16h* bp = (const v16h*)Bp;

  v8f acc[TN] = {};
#pragma unroll
  for (int kc = 0; kc < KC; ++kc) {
    int kb = kc * 32 + hi * 8;
    v4f a0 = *(const v4f*)(arow + kb);
    v4f a1 = *(const v4f*)(arow + kb + 4);
    v4f a2 = *(const v4f*)(arow + kb + 16);
    v4f a3 = *(const v4f*)(arow + kb + 20);
    v16h a;
#pragma unroll
    for (int e = 0; e < 4; ++e) {
      float x0 = a0[e], x1 = a1[e], x2 = a2[e], x3 = a3[e];
      if (ELU_A) { x0 = elu_f(x0); x1 = elu_f(x1); x2 = elu_f(x2); x3 = elu_f(x3); }
      a[e]      = (_Float16)x0;
      a[4 + e]  = (_Float16)x1;
      a[8 + e]  = (_Float16)x2;
      a[12 + e] = (_Float16)x3;
    }
#pragma unroll
    for (int tn = 0; tn < TN; ++tn) {
      v16h b = bp[(size_t)(tn * KC + kc) * 32 + lane];
      acc[tn] = __builtin_amdgcn_wmma_f32_16x16x32_f16(
          /*neg_a=*/false, a, /*neg_b=*/false, b,
          /*c_mod=*/(short)0, acc[tn], /*reuse_a=*/false, /*reuse_b=*/false);
    }
  }
  // C/D layout: VGPR v -> row = tm*16 + hi*8 + v, col = tn*16 + ml
#pragma unroll
  for (int tn = 0; tn < TN; ++tn) {
    float* yp = Y + (size_t)(tm * 16 + hi * 8) * NC + tn * 16 + ml;
#pragma unroll
    for (int v = 0; v < 8; ++v) yp[(size_t)v * NC] = acc[tn][v];
  }
}

// ---------------- sparse aggregate: wave per edge, f32 atomics (L2-resident) ----
template<int H>
__global__ void spmm_atomic(const int* __restrict__ src, const int* __restrict__ dst,
                            const float* __restrict__ att, const float* __restrict__ Y,
                            float* __restrict__ Z) {
  int e = blockIdx.x * (blockDim.x >> 5) + (threadIdx.x >> 5);
  if (e >= NEDGES) return;
  int lane = threadIdx.x & 31;
  int s = src[e], d = dst[e];
  float a = att[e];
  const float* y = Y + (size_t)d * H;
  float* z = Z + (size_t)s * H;
#pragma unroll
  for (int c = lane; c < H; c += 32) atomicAdd(&z[c], a * y[c]);
}

// ---------------- final ELU + log_softmax: NCLS==32 -> 1 class per lane ----------
__global__ void log_softmax32(const float* __restrict__ Z, float* __restrict__ out) {
  int node = blockIdx.x * (blockDim.x >> 5) + (threadIdx.x >> 5);
  if (node >= NNODES) return;
  int lane = threadIdx.x & 31;
  float x = elu_f(Z[(size_t)node * NCLS + lane]);
  float m = wave_max(x);
  float ex = expf(x - m);
  float s = wave_sum(ex);
  out[(size_t)node * NCLS + lane] = x - m - logf(s);
}

// ---------------- orchestration ----------------
extern "C" void kernel_launch(void* const* d_in, const int* in_sizes, int n_in,
                              void* d_out, int out_size, void* d_ws, size_t ws_size,
                              hipStream_t stream) {
  const float* features = (const float*)d_in[0];
  const float* hidden   = (const float*)d_in[1];
  const int*   adj      = (const int*)d_in[2];
  const float* W0       = (const float*)d_in[3];
  const float* W1       = (const float*)d_in[4];
  const float* W2       = (const float*)d_in[5];
  const float* attw     = (const float*)d_in[6];
  const int* src = adj;
  const int* dst = adj + NEDGES;
  float* out = (float*)d_out;

  // workspace carve-out; all offsets multiples of 32B
  float* ws     = (float*)d_ws;
  float* pa     = ws;                          // N
  float* pb     = pa + NNODES;                 // N
  float* rowmax = pb + NNODES;                 // N
  float* rowsum = rowmax + NNODES;             // N
  float* att    = rowsum + NNODES;             // E
  float* Y      = att + NEDGES;                // N*HIDDIM
  float* Z      = Y + (size_t)NNODES * HIDDIM; // N*HIDDIM
  _Float16* bp0 = (_Float16*)(Z + (size_t)NNODES * HIDDIM);      // 32*32*16 halves
  _Float16* bp1 = bp0 + (size_t)(INDIM / 32) * (HIDDIM / 16) * 32 * 16;
  _Float16* bp2 = bp1 + (size_t)(HIDDIM / 32) * (HIDDIM / 16) * 32 * 16;

  dim3 blk(256);

  // --- weight pre-pack into WMMA B-fragment layout (tiny, once per call) ---
  pack_w<INDIM, HIDDIM><<<4, blk, 0, stream>>>(W0, bp0);   // 1024 threads
  pack_w<HIDDIM, HIDDIM><<<1, blk, 0, stream>>>(W1, bp1);  // 256 threads
  pack_w<HIDDIM, NCLS><<<1, blk, 0, stream>>>(W2, bp2);    // 128 threads

  // --- attention ---
  proj_kernel<<<(NNODES + 7) / 8, blk, 0, stream>>>(hidden, attw, pa, pb);
  init_rows<<<(NNODES + 255) / 256, blk, 0, stream>>>(rowmax, rowsum);
  edge_score<<<(NEDGES + 255) / 256, blk, 0, stream>>>(src, dst, pa, pb, att, rowmax);
  edge_exp<<<(NEDGES + 255) / 256, blk, 0, stream>>>(src, att, rowmax, rowsum);
  edge_norm<<<(NEDGES + 255) / 256, blk, 0, stream>>>(src, att, rowsum);

  const int nh = NNODES * HIDDIM;
  const int nc = NNODES * NCLS;
  const int strip_blocks = (NNODES / 16 + 7) / 8;   // one wave per 16-row strip

  // --- layer 1: Y = features @ W0 ; Z = spmm(att, Y) ---
  gemm_wmma<INDIM, HIDDIM, false><<<strip_blocks, blk, 0, stream>>>(features, bp0, Y);
  zero_kernel<<<(nh + 255) / 256, blk, 0, stream>>>(Z, nh);
  spmm_atomic<HIDDIM><<<(NEDGES + 7) / 8, blk, 0, stream>>>(src, dst, att, Y, Z);

  // --- layer 2: Y = elu(Z) @ W1 ; Z = spmm(att, Y)  (ELU fused into A-read) ---
  gemm_wmma<HIDDIM, HIDDIM, true><<<strip_blocks, blk, 0, stream>>>(Z, bp1, Y);
  zero_kernel<<<(nh + 255) / 256, blk, 0, stream>>>(Z, nh);
  spmm_atomic<HIDDIM><<<(NEDGES + 7) / 8, blk, 0, stream>>>(src, dst, att, Y, Z);

  // --- layer 3: Y = elu(Z) @ W2 ; Z = spmm(att, Y) ; out = log_softmax(elu(Z)) ---
  gemm_wmma<HIDDIM, NCLS, true><<<strip_blocks, blk, 0, stream>>>(Z, bp2, Y);
  zero_kernel<<<(nc + 255) / 256, blk, 0, stream>>>(Z, nc);
  spmm_atomic<NCLS><<<(NEDGES + 7) / 8, blk, 0, stream>>>(src, dst, att, Y, Z);
  log_softmax32<<<(NNODES + 7) / 8, blk, 0, stream>>>(Z, out);
}
